// OctreeDeConvBnRelu_8624294331030
// MI455X (gfx1250) — matrix operand order repaired and checked by reading, not compiled, not run on hardware
//
#include <hip/hip_runtime.h>
#include <hip/hip_bf16.h>

// OctreeDeconv(27-tap) + BatchNorm + ReLU for MI455X (gfx1250, wave32).
// GEMM path: v_wmma_f32_16x16x32_bf16 (fp32 accumulation), scatter via
// L2-resident global_atomic_add_f32.

#define NODES_DEFAULT 800000
#define CIN  64
#define COUT 64
#define KTAP 27
#define BN_EPS 1e-5f

typedef __attribute__((ext_vector_type(16))) __bf16        v16bf;
typedef __attribute__((ext_vector_type(8)))  float         v8f;
typedef __attribute__((ext_vector_type(8)))  unsigned int  v8u;

__device__ __forceinline__ unsigned short f2bf(float f) {
  union { float f; unsigned u; } v; v.f = f;
  unsigned u = v.u;
  unsigned r = u + 0x7FFFu + ((u >> 16) & 1u);   // round-to-nearest-even
  return (unsigned short)(r >> 16);
}

// ---------------------------------------------------------------------------
// Kernel 0: zero accumulator + stats, convert W[k][cin][cout] fp32 ->
//           Wt[k][cout][cin] bf16 (transposed so B fragments read contiguous cin)
// ---------------------------------------------------------------------------
__global__ __launch_bounds__(256)
void od_prep(const float* __restrict__ w, float4* __restrict__ acc4,
             unsigned short* __restrict__ wt, float* __restrict__ stats,
             int n_nodes) {
  size_t gid = (size_t)blockIdx.x * blockDim.x + threadIdx.x;
  size_t tot4 = (size_t)n_nodes * COUT / 4;
  if (gid < tot4) acc4[gid] = make_float4(0.f, 0.f, 0.f, 0.f);
  if (gid < (size_t)KTAP * CIN * COUT) {
    int k   = (int)(gid / (CIN * COUT));
    int rem = (int)(gid % (CIN * COUT));
    int co  = rem >> 6, ci = rem & 63;
    wt[gid] = f2bf(w[(size_t)k * CIN * COUT + (size_t)ci * COUT + co]);
  }
  if (gid < 2 * COUT) stats[gid] = 0.f;
}

// ---------------------------------------------------------------------------
// Kernel 1: GEMM (bf16 WMMA, fp32 acc) + neigh scatter (f32 atomics into ws)
//   block = 256 threads (8 waves), tile = 128 nodes; wave w owns nodes
//   [16w, 16w+16). Per k: 4 cout tiles x 2 WMMAs (K=64 = 2x32).
// ---------------------------------------------------------------------------
#define TM 128
#define ROWP 36   // LDS row pitch in dwords (72 bf16 = 144 B, 16B aligned)

__global__ __launch_bounds__(256)
void od_gemm_scatter(const float* __restrict__ data,      // [CIN][N] fp32
                     const int*   __restrict__ neigh,     // [N][27]
                     const unsigned short* __restrict__ wt,// [27][COUT][CIN] bf16
                     float* __restrict__ acc,             // [N][COUT] fp32
                     int n_nodes) {
  __shared__ unsigned int xlds[TM * ROWP];   // x tile, [node][cin] bf16
  __shared__ unsigned int wlds[COUT * ROWP]; // W[k]^T tile, [cout][cin] bf16

  const int tid = threadIdx.x;
  const int n0  = blockIdx.x * TM;

  // --- stage x tile: coalesced along nodes, convert fp32 -> bf16 ---
  {
    unsigned short* xs = (unsigned short*)xlds;
    for (int i = tid; i < TM * CIN; i += 256) {
      int c = i >> 7;        // channel
      int m = i & 127;       // node within tile
      xs[m * (2 * ROWP) + c] = f2bf(data[(size_t)c * n_nodes + (n0 + m)]);
    }
  }
  __syncthreads();

  const int lane  = tid & 31;
  const int wave  = tid >> 5;
  const int mrow  = lane & 15;   // A: row m / B,D: col n
  const int half  = lane >> 4;
  const int mbase = wave * 16;

  // --- A fragments per ISA table: elem e<8 -> K=half*8+e, e>=8 -> K=16+half*8+(e-8)
  v8u a0w, a1w;
  {
    const unsigned int* xr = xlds + (mbase + mrow) * ROWP;
#pragma unroll
    for (int t = 0; t < 4; ++t) {
      a0w[t]     = xr[half * 4 + t];            // K  0..15 region
      a0w[t + 4] = xr[8 + half * 4 + t];        // K 16..31 region
      a1w[t]     = xr[16 + half * 4 + t];       // K 32..47 region
      a1w[t + 4] = xr[24 + half * 4 + t];       // K 48..63 region
    }
  }
  const v16bf a0 = __builtin_bit_cast(v16bf, a0w);
  const v16bf a1 = __builtin_bit_cast(v16bf, a1w);

  for (int k = 0; k < KTAP; ++k) {
    __syncthreads();
    // --- stage W[k]^T (already transposed bf16 in global) ---
    {
      unsigned short* wsh = (unsigned short*)wlds;
      const unsigned short* wg = wt + (size_t)k * CIN * COUT;
      for (int i = tid; i < CIN * COUT; i += 256) {
        int co = i >> 6, ci = i & 63;
        wsh[co * (2 * ROWP) + ci] = wg[i];
      }
    }
    __syncthreads();

    // destination nodes for this lane's 8 D rows (m = half*8 + r)
    int dst[8];
#pragma unroll
    for (int r = 0; r < 8; ++r)
      dst[r] = neigh[(size_t)(n0 + mbase + half * 8 + r) * KTAP + k];

#pragma unroll
    for (int jt = 0; jt < 4; ++jt) {
      // B fragments: elem e<8 -> K = half*8 + e, e>=8 -> K = 16 + half*8 + (e-8)
      const unsigned int* wr = wlds + (jt * 16 + mrow) * ROWP;
      v8u b0w, b1w;
#pragma unroll
      for (int t = 0; t < 8; ++t) {
        b0w[t] = wr[(t >> 2) * 8 + half * 4 + (t & 3)];        // cin  0..31
        b1w[t] = wr[16 + (t >> 2) * 8 + half * 4 + (t & 3)];   // cin 32..63
      }
      v8f c = {};
      c = __builtin_amdgcn_wmma_f32_16x16x32_bf16(
              false, a0, false, __builtin_bit_cast(v16bf, b0w),
              (short)0, c, false, false);
      c = __builtin_amdgcn_wmma_f32_16x16x32_bf16(
              false, a1, false, __builtin_bit_cast(v16bf, b1w),
              (short)0, c, false, false);
      // scatter D: VGPR r holds (m = half*8+r, n = mrow)
#pragma unroll
      for (int r = 0; r < 8; ++r) {
        float* p = acc + (size_t)dst[r] * COUT + jt * 16 + mrow;
        __hip_atomic_fetch_add(p, c[r], __ATOMIC_RELAXED,
                               __HIP_MEMORY_SCOPE_AGENT);
      }
    }
  }
}

// ---------------------------------------------------------------------------
// Kernel 2: per-channel sum / sum-of-squares reduction
// ---------------------------------------------------------------------------
__global__ __launch_bounds__(256)
void od_bn_stats(const float* __restrict__ acc, float* __restrict__ stats,
                 int n_nodes) {
  __shared__ float ssum[256], ssq[256];
  const int tid = threadIdx.x;
  const int c   = tid & 63;
  const int r   = tid >> 6;
  size_t base = (size_t)blockIdx.x * 1024;
  size_t end  = base + 1024; if (end > (size_t)n_nodes) end = (size_t)n_nodes;
  float s = 0.f, q = 0.f;
  for (size_t n = base + r; n < end; n += 4) {
    float v = acc[n * COUT + c];
    s += v; q += v * v;
  }
  ssum[tid] = s; ssq[tid] = q;
  __syncthreads();
  if (tid < 64) {
    float ts = ssum[tid] + ssum[tid + 64] + ssum[tid + 128] + ssum[tid + 192];
    float tq = ssq[tid]  + ssq[tid + 64]  + ssq[tid + 128]  + ssq[tid + 192];
    __hip_atomic_fetch_add(stats + tid,       ts, __ATOMIC_RELAXED,
                           __HIP_MEMORY_SCOPE_AGENT);
    __hip_atomic_fetch_add(stats + 64 + tid,  tq, __ATOMIC_RELAXED,
                           __HIP_MEMORY_SCOPE_AGENT);
  }
}

// ---------------------------------------------------------------------------
// Kernel 3: normalize + ReLU + transpose to [COUT][N] (LDS 64x64 transpose)
// ---------------------------------------------------------------------------
__global__ __launch_bounds__(256)
void od_bn_apply(const float* __restrict__ acc, const float* __restrict__ stats,
                 const float* __restrict__ gamma, const float* __restrict__ beta,
                 float* __restrict__ out, int n_nodes) {
  __shared__ float T[64 * 65];
  __shared__ float sc[64], sh[64];
  const int tid = threadIdx.x;
  const int n0  = blockIdx.x * 64;
  if (tid < 64) {
    float invn = 1.0f / (float)n_nodes;
    float mean = stats[tid] * invn;
    float var  = stats[64 + tid] * invn - mean * mean;
    float g    = gamma[tid] * rsqrtf(var + BN_EPS);
    sc[tid] = g;
    sh[tid] = beta[tid] - mean * g;
  }
  __syncthreads();
#pragma unroll
  for (int j = 0; j < 16; ++j) {
    int idx = j * 256 + tid;
    int i = idx >> 6, c = idx & 63;
    float v = acc[(size_t)(n0 + i) * COUT + c];   // coalesced read
    v = v * sc[c] + sh[c];
    T[c * 65 + i] = v > 0.f ? v : 0.f;            // ReLU
  }
  __syncthreads();
#pragma unroll
  for (int j = 0; j < 16; ++j) {
    int i = tid & 63;
    int c = (tid >> 6) + 4 * j;
    out[(size_t)c * n_nodes + (n0 + i)] = T[c * 65 + i];  // coalesced write
  }
}

// ---------------------------------------------------------------------------
extern "C" void kernel_launch(void* const* d_in, const int* in_sizes, int n_in,
                              void* d_out, int out_size, void* d_ws, size_t ws_size,
                              hipStream_t stream) {
  const float* data   = (const float*)d_in[0];   // [1,CIN,N,1]
  const int*   neigh  = (const int*)d_in[1];     // [N,27]
  const float* weight = (const float*)d_in[2];   // [27,CIN,COUT]
  const float* gamma  = (const float*)d_in[3];   // [COUT]
  const float* beta   = (const float*)d_in[4];   // [COUT]
  float*       out    = (float*)d_out;           // [1,COUT,N,1]

  const int n_nodes = in_sizes[1] / KTAP;        // 800000 (divisible by 128)

  char* ws = (char*)d_ws;
  float*          acc   = (float*)ws;                                  // N*64 f32
  unsigned short* wt    = (unsigned short*)(ws + (size_t)n_nodes * COUT * 4);
  float*          stats = (float*)(ws + (size_t)n_nodes * COUT * 4
                                      + (size_t)KTAP * CIN * COUT * 2);

  size_t tot4 = (size_t)n_nodes * COUT / 4;
  int pblocks = (int)((tot4 + 255) / 256);
  od_prep<<<pblocks, 256, 0, stream>>>(weight, (float4*)acc, wt, stats, n_nodes);

  od_gemm_scatter<<<n_nodes / TM, 256, 0, stream>>>(data, neigh, wt, acc, n_nodes);

  od_bn_stats<<<(n_nodes + 1023) / 1024, 256, 0, stream>>>(acc, stats, n_nodes);

  od_bn_apply<<<n_nodes / 64, 256, 0, stream>>>(acc, stats, gamma, beta, out,
                                                n_nodes);
}
